// VertexConv_48034914239356
// MI455X (gfx1250) — matrix operand order repaired and checked by reading, not compiled, not run on hardware
//
#include <hip/hip_runtime.h>

typedef __attribute__((ext_vector_type(16))) __bf16 v16bf;
typedef __attribute__((ext_vector_type(8)))  __bf16 v8bf;
typedef __attribute__((ext_vector_type(8)))  float  v8f;

#define M_NODES 8192
#define E_EDGES 2048
#define D_FEAT  256

// Tile geometry shared by all GEMMs: 64(M) x 128(N) x 32(K), 256 thr = 8 waves,
// waves 2(M) x 4(N), each wave owns 32x32 output (2 A-frags x 2 B-frags).
#define BM 64
#define BN 128
#define BK 32
#define LDK 40   // BK + 8 bf16 pad (80B row stride, 16B-aligned chunks)

// ---------------------------------------------------------------------------
// Fragment loads: LDS tiles are K-fastest per row, so per-lane fragment data is
// two contiguous 16B runs -> 2 x ds_load_b128 each (CDNA5 layouts, CDNA5 ISA).
// A 16x32 bf16: lane L: M=L&15, h=L>>4; elements = K[8h..8h+7] ++ K[16+8h..23+8h]
__device__ __forceinline__ v16bf load_a_frag(const __bf16* As, int mbase, int lane) {
  const __bf16* p = As + (size_t)(mbase + (lane & 15)) * LDK + ((lane >> 4) & 1) * 8;
  v8bf lo = *(const v8bf*)(p);
  v8bf hi = *(const v8bf*)(p + 16);
  return __builtin_shufflevector(lo, hi, 0,1,2,3,4,5,6,7,8,9,10,11,12,13,14,15);
}
// B 32x16 bf16: lane L: N=L&15, h=L>>4; elements = K[8h..8h+7] ++ K[16+8h..23+8h]
__device__ __forceinline__ v16bf load_b_frag(const __bf16* Bs, int nbase, int lane) {
  const __bf16* p = Bs + (size_t)(nbase + (lane & 15)) * LDK + ((lane >> 4) & 1) * 8;
  v8bf lo = *(const v8bf*)(p);
  v8bf hi = *(const v8bf*)(p + 16);
  return __builtin_shufflevector(lo, hi, 0,1,2,3,4,5,6,7,8,9,10,11,12,13,14,15);
}

#define WMMA_BF16(acc, a, b) \
  acc = __builtin_amdgcn_wmma_f32_16x16x32_bf16(false, a, false, b, (short)0, acc, false, false)

// ---------------------------------------------------------------------------
// f32 -> bf16, vectorized (element count multiple of 1024; exact here)
__global__ __launch_bounds__(256) void k_cvt_bf16_v4(const float* __restrict__ in,
                                                     __bf16* __restrict__ out) {
  int i = blockIdx.x * 256 + threadIdx.x;
  float4 v = ((const float4*)in)[i];
  alignas(8) __bf16 p[4] = {(__bf16)v.x, (__bf16)v.y, (__bf16)v.z, (__bf16)v.w};
  ((uint2*)out)[i] = *(uint2*)p;
}

// feat [M,D] f32 -> featT [D,M] bf16, LDS-tiled 64x64 transpose
__global__ __launch_bounds__(256) void k_transpose_feat(const float* __restrict__ in,
                                                        __bf16* __restrict__ outT) {
  __shared__ alignas(16) __bf16 T[64][72];
  const int tid = threadIdx.x;
  const int m0 = blockIdx.x * 64, d0 = blockIdx.y * 64;
#pragma unroll
  for (int it = 0; it < 4; ++it) {
    int idx = tid + it * 256;                 // 1024 float4 chunks
    int m = idx >> 4, d4 = (idx & 15) * 4;
    float4 v = *(const float4*)(in + (size_t)(m0 + m) * D_FEAT + d0 + d4);
    alignas(8) __bf16 p[4] = {(__bf16)v.x, (__bf16)v.y, (__bf16)v.z, (__bf16)v.w};
    *(uint2*)&T[m][d4] = *(uint2*)p;
  }
  __syncthreads();
#pragma unroll
  for (int it = 0; it < 4; ++it) {
    int idx = tid + it * 256;
    int d = idx >> 4, m4 = (idx & 15) * 4;
    alignas(8) __bf16 p[4] = {T[m4][d], T[m4 + 1][d], T[m4 + 2][d], T[m4 + 3][d]};
    *(uint2*)(outT + (size_t)(d0 + d) * M_NODES + m0 + m4) = *(uint2*)p;
  }
}

// ---------------------------------------------------------------------------
// "TN" bf16 GEMM: C[M,N] = A[M,K] @ B[N,K]^T; A row-major [m][k], B row-major
// [n][k] (both K-contiguous -> pure b128 fills). Double-buffered LDS pipeline:
// one barrier per K-step, global loads overlap the WMMA phase.
// MODE 0: store C transposed as bf16 CT[N, MtotT] (packed b128 stores).
// MODE 1: out[m*Ndim+n] = alpha*prev + (1-alpha)*acc, f32.
template <int MODE>
__global__ __launch_bounds__(256)
void k_gemm(const __bf16* __restrict__ A, const __bf16* __restrict__ B,
            void* __restrict__ Cout, const float* __restrict__ prev,
            const float* __restrict__ alphaPtr, int Kdim, int Ndim, int MtotT) {
  __shared__ alignas(16) __bf16 As[2][BM * LDK];
  __shared__ alignas(16) __bf16 Bs[2][BN * LDK];
  const int tid = threadIdx.x, lane = tid & 31, wid = tid >> 5;
  const int wm = wid & 1, wn = wid >> 1;          // 2 x 4 waves
  const int m0 = blockIdx.x * BM, n0 = blockIdx.y * BN;
  // per-thread fill coordinates
  const int am = tid >> 2, k8 = (tid & 3) * 8;    // A: 256 x 16B chunks
  const int bn = tid >> 2;                        // B: 512 chunks (2 its)
  v8f acc[2][2] = {};
  uint4 avv, bvv[2];

  // prologue: fill buffer 0 from k0 = 0
  avv = *(const uint4*)(A + (size_t)(m0 + am) * Kdim + k8);
#pragma unroll
  for (int it = 0; it < 2; ++it)
    bvv[it] = *(const uint4*)(B + (size_t)(n0 + bn + it * 64) * Kdim + k8);
  *(uint4*)&As[0][am * LDK + k8] = avv;
#pragma unroll
  for (int it = 0; it < 2; ++it)
    *(uint4*)&Bs[0][(bn + it * 64) * LDK + k8] = bvv[it];
  __syncthreads();

  int buf = 0;
  for (int k0 = 0; k0 < Kdim; k0 += BK) {
    const bool nxt = (k0 + BK) < Kdim;
    if (nxt) {
      avv = *(const uint4*)(A + (size_t)(m0 + am) * Kdim + k0 + BK + k8);
#pragma unroll
      for (int it = 0; it < 2; ++it)
        bvv[it] = *(const uint4*)(B + (size_t)(n0 + bn + it * 64) * Kdim + k0 + BK + k8);
    }
    v16bf a0 = load_a_frag(As[buf], wm * 32, lane);
    v16bf a1 = load_a_frag(As[buf], wm * 32 + 16, lane);
    v16bf b0 = load_b_frag(Bs[buf], wn * 32, lane);
    v16bf b1 = load_b_frag(Bs[buf], wn * 32 + 16, lane);
    WMMA_BF16(acc[0][0], a0, b0);
    WMMA_BF16(acc[0][1], a0, b1);
    WMMA_BF16(acc[1][0], a1, b0);
    WMMA_BF16(acc[1][1], a1, b1);
    if (nxt) {
      *(uint4*)&As[buf ^ 1][am * LDK + k8] = avv;
#pragma unroll
      for (int it = 0; it < 2; ++it)
        *(uint4*)&Bs[buf ^ 1][(bn + it * 64) * LDK + k8] = bvv[it];
    }
    __syncthreads();
    buf ^= 1;
  }

  const int h = (lane >> 4) & 1;
  if (MODE == 0) {      // CT[n][m] bf16; D rows are M-consecutive -> b128 pack
    __bf16* CT = (__bf16*)Cout;
#pragma unroll
    for (int af = 0; af < 2; ++af)
#pragma unroll
      for (int bf = 0; bf < 2; ++bf) {
        int n = n0 + wn * 32 + bf * 16 + (lane & 15);
        int mb = m0 + wm * 32 + af * 16 + 8 * h;
        v8bf p;
#pragma unroll
        for (int r = 0; r < 8; ++r) p[r] = (__bf16)acc[af][bf][r];
        *(v8bf*)(CT + (size_t)n * MtotT + mb) = p;
      }
  } else {
    float* C = (float*)Cout;
    const float a0v = *alphaPtr, b0v = 1.0f - a0v;
#pragma unroll
    for (int af = 0; af < 2; ++af)
#pragma unroll
      for (int bf = 0; bf < 2; ++bf) {
        int n = n0 + wn * 32 + bf * 16 + (lane & 15);
        int mb = m0 + wm * 32 + af * 16 + 8 * h;
#pragma unroll
        for (int r = 0; r < 8; ++r) {
          size_t idx = (size_t)(mb + r) * Ndim + n;
          C[idx] = a0v * prev[idx] + b0v * acc[af][bf][r];
        }
      }
  }
}

// ---------------------------------------------------------------------------
// Fused big GEMM: ESAG[e][n] = sum_m inc[m][e] * X[m][n], n in [0,512),
// X = concat(node_scores | feature). inc (64 MB, the dominant HBM stream) is
// read once, coalesced along e, transposed into As[e][k] via packed-cvt LDS
// stores. B comes pre-transposed (nsT/featT are [feature][node]) -> b128 fills.
// Double-buffered LDS pipeline + L2 prefetch 8 tiles ahead on the inc stream.
__global__ __launch_bounds__(256)
void k_gemm_big(const float* __restrict__ inc, const __bf16* __restrict__ nsT,
                const __bf16* __restrict__ featT, float* __restrict__ ESAG) {
  __shared__ alignas(16) __bf16 As[2][BM * LDK];
  __shared__ alignas(16) __bf16 Bs[2][BN * LDK];
  const int tid = threadIdx.x, lane = tid & 31, wid = tid >> 5;
  const int wm = wid & 1, wn = wid >> 1;
  const int e0 = blockIdx.x * BM;
  const int n0 = blockIdx.y * BN;             // 0,128,256,384
  const __bf16* Xb = (n0 < D_FEAT) ? (nsT + (size_t)n0 * M_NODES)
                                   : (featT + (size_t)(n0 - D_FEAT) * M_NODES);
  // per-thread fill coordinates
  const int ak = tid >> 4, e4 = (tid & 15) * 4;   // A: 512 float4 chunks (2 its)
  const int bn = tid >> 2, k8 = (tid & 3) * 8;    // B: 512 x 16B chunks (2 its)
  v8f acc[2][2] = {};
  float4 av[2];
  uint4 bv[2];

  // prologue: fill buffer 0 from k0 = 0
#pragma unroll
  for (int it = 0; it < 2; ++it) {
    av[it] = *(const float4*)(inc + (size_t)(ak + it * 16) * E_EDGES + e0 + e4);
    bv[it] = *(const uint4*)(Xb + (size_t)(bn + it * 64) * M_NODES + k8);
  }
#pragma unroll
  for (int it = 0; it < 2; ++it) {
    __bf16* dst = &As[0][e4 * LDK + ak + it * 16];
    dst[0 * LDK] = (__bf16)av[it].x;
    dst[1 * LDK] = (__bf16)av[it].y;
    dst[2 * LDK] = (__bf16)av[it].z;
    dst[3 * LDK] = (__bf16)av[it].w;
    *(uint4*)&Bs[0][(bn + it * 64) * LDK + k8] = bv[it];
  }
  __syncthreads();

  int buf = 0;
  for (int k0 = 0; k0 < M_NODES; k0 += BK) {
    const bool nxt = (k0 + BK) < M_NODES;
    if (nxt) {
#pragma unroll
      for (int it = 0; it < 2; ++it) {
        av[it] = *(const float4*)(inc + (size_t)(k0 + BK + ak + it * 16) * E_EDGES + e0 + e4);
        bv[it] = *(const uint4*)(Xb + (size_t)(bn + it * 64) * M_NODES + k0 + BK + k8);
      }
    }
    if (k0 + 8 * BK < M_NODES)  // warm L2 ahead on the 64MB inc stream
      __builtin_prefetch(inc + (size_t)(k0 + 8 * BK + ak) * E_EDGES + e0 + e4, 0, 1);

    v16bf a0 = load_a_frag(As[buf], wm * 32, lane);
    v16bf a1 = load_a_frag(As[buf], wm * 32 + 16, lane);
    v16bf b0 = load_b_frag(Bs[buf], wn * 32, lane);
    v16bf b1 = load_b_frag(Bs[buf], wn * 32 + 16, lane);
    WMMA_BF16(acc[0][0], a0, b0);
    WMMA_BF16(acc[0][1], a0, b1);
    WMMA_BF16(acc[1][0], a1, b0);
    WMMA_BF16(acc[1][1], a1, b1);

    if (nxt) {
#pragma unroll
      for (int it = 0; it < 2; ++it) {
        __bf16* dst = &As[buf ^ 1][e4 * LDK + ak + it * 16];
        dst[0 * LDK] = (__bf16)av[it].x;
        dst[1 * LDK] = (__bf16)av[it].y;
        dst[2 * LDK] = (__bf16)av[it].z;
        dst[3 * LDK] = (__bf16)av[it].w;
        *(uint4*)&Bs[buf ^ 1][(bn + it * 64) * LDK + k8] = bv[it];
      }
    }
    __syncthreads();
    buf ^= 1;
  }

  const int h = (lane >> 4) & 1;
#pragma unroll
  for (int af = 0; af < 2; ++af)
#pragma unroll
    for (int bf = 0; bf < 2; ++bf) {
      int n = n0 + wn * 32 + bf * 16 + (lane & 15);
      int eb = e0 + wm * 32 + af * 16 + 8 * h;
#pragma unroll
      for (int r = 0; r < 8; ++r)
        ESAG[(size_t)(eb + r) * 512 + n] = acc[af][bf][r];
    }
}

// ---------------------------------------------------------------------------
// Softmax over D per edge + elementwise mul with agg -> EF bf16 [E, D]
__global__ __launch_bounds__(256)
void k_softmax_mul(const float* __restrict__ ESAG, __bf16* __restrict__ EF) {
  __shared__ float red[256];
  const int e = blockIdx.x, d = threadIdx.x;
  const float s = ESAG[(size_t)e * 512 + d];
  red[d] = s;
  __syncthreads();
#pragma unroll
  for (int off = 128; off > 0; off >>= 1) {
    if (d < off) red[d] = fmaxf(red[d], red[d + off]);
    __syncthreads();
  }
  const float mx = red[0];
  __syncthreads();
  const float ex = __expf(s - mx);
  red[d] = ex;
  __syncthreads();
#pragma unroll
  for (int off = 128; off > 0; off >>= 1) {
    if (d < off) red[d] += red[d + off];
    __syncthreads();
  }
  const float attw = ex / red[0];
  const float agg = ESAG[(size_t)e * 512 + 256 + d];
  EF[(size_t)e * D_FEAT + d] = (__bf16)(agg * attw);
}

// ---------------------------------------------------------------------------
extern "C" void kernel_launch(void* const* d_in, const int* in_sizes, int n_in,
                              void* d_out, int out_size, void* d_ws, size_t ws_size,
                              hipStream_t stream) {
  const float* feat  = (const float*)d_in[0];   // [M, D]
  const float* inc   = (const float*)d_in[1];   // [M, E]
  const float* prev  = (const float*)d_in[2];   // [E, D]
  const float* W_att = (const float*)d_in[3];   // [D, D] = [n][k]
  const float* W_prj = (const float*)d_in[4];   // [D, D] = [n][k]
  const float* alpha = (const float*)d_in[5];   // [1]

  // workspace layout (~17.3 MB)
  __bf16* featB = (__bf16*)d_ws;                        // [M,D] row-major
  __bf16* featT = featB + (size_t)M_NODES * D_FEAT;     // [D,M]
  __bf16* nsT   = featT + (size_t)M_NODES * D_FEAT;     // [D,M]
  __bf16* WattB = nsT   + (size_t)M_NODES * D_FEAT;     // [D,D] = [n][k]
  __bf16* WprjB = WattB + (size_t)D_FEAT * D_FEAT;      // [D,D]
  float*  ESAG  = (float*)(WprjB + (size_t)D_FEAT * D_FEAT); // [E,512]
  __bf16* EF    = (__bf16*)(ESAG + (size_t)E_EDGES * 512);   // [E,D]

  // 1. conversions (weights already in B-friendly [n][k] layout)
  k_cvt_bf16_v4<<<(M_NODES * D_FEAT) / 1024, 256, 0, stream>>>(feat, featB);
  k_cvt_bf16_v4<<<(D_FEAT * D_FEAT) / 1024, 256, 0, stream>>>(W_att, WattB);
  k_cvt_bf16_v4<<<(D_FEAT * D_FEAT) / 1024, 256, 0, stream>>>(W_prj, WprjB);
  k_transpose_feat<<<dim3(M_NODES / 64, D_FEAT / 64), 256, 0, stream>>>(feat, featT);

  // 2. nsT[D,M] = (featB @ W_att^T)^T, bf16 (transposed for big-GEMM B operand)
  k_gemm<0><<<dim3(M_NODES / BM, D_FEAT / BN), 256, 0, stream>>>(
      featB, WattB, (void*)nsT, nullptr, nullptr, D_FEAT, D_FEAT, M_NODES);

  // 3. fused inc^T @ [node_scores | feature] -> ESAG f32 [E,512]
  k_gemm_big<<<dim3(E_EDGES / BM, 512 / BN), 256, 0, stream>>>(inc, nsT, featT, ESAG);

  // 4. softmax over D + mul -> EF bf16 [E,D]
  k_softmax_mul<<<E_EDGES, 256, 0, stream>>>(ESAG, EF);

  // 5. out = alpha*prev + (1-alpha) * (EF @ W_proj^T), f32
  k_gemm<1><<<dim3(E_EDGES / BM, D_FEAT / BN), 256, 0, stream>>>(
      EF, WprjB, d_out, prev, alpha, D_FEAT, D_FEAT, 0);
}
